// DeformableAttention_68418829025655
// MI455X (gfx1250) — matrix operand order repaired and checked by reading, not compiled, not run on hardware
//
#include <hip/hip_runtime.h>
#include <hip/hip_bf16.h>

// ---------------------------------------------------------------------------
// Problem constants: B=4, N=16384, NV=16384, D=256, NH=8, P=4, HD=32
// ---------------------------------------------------------------------------
#define KDIM   256
#define MROWS  65536   // B*N == B*NV
#define NHEAD  8
#define NPTS   4
#define HDIM   32
#define HW     128     // sqrt(NV) feature map side

typedef __bf16 v16bf __attribute__((ext_vector_type(16)));
typedef float  v8f   __attribute__((ext_vector_type(8)));

union Frag16 {
    float4 f4[2];
    v16bf  v;
};

__device__ __forceinline__ unsigned short bf16_rne(float x) {
    unsigned int u = __float_as_uint(x);
    unsigned int r = u + 0x7FFFu + ((u >> 16) & 1u);
    return (unsigned short)(r >> 16);
}
__device__ __forceinline__ float bf16_to_f(unsigned short h) {
    return __uint_as_float(((unsigned int)h) << 16);
}

// ---------------------------------------------------------------------------
// Tensor Data Mover (TDM) support for the weight-tile staging.
// ---------------------------------------------------------------------------
#if __has_builtin(__builtin_amdgcn_tensor_load_to_lds) && \
    __has_builtin(__builtin_amdgcn_s_wait_tensorcnt)
#define USE_TDM 1
#else
#define USE_TDM 0
#endif

#if USE_TDM
typedef unsigned int tdm_u32x4 __attribute__((ext_vector_type(4)));
typedef int          tdm_i32x4 __attribute__((ext_vector_type(4)));
typedef int          tdm_i32x8 __attribute__((ext_vector_type(8)));

// DMA a 2D bf16 tile [128 rows x 32 elems] from a row-major [Ncols][256]
// bf16 tensor into LDS, with hardware padding of 4 DWORDs after every
// 16 DWORDs (reproducing the 80B LDS row stride used by the fragment loads).
__device__ __forceinline__ void tdm_load_tile(
    const unsigned short* gsrc, unsigned lds_off, int n_rows_tensor)
{
    unsigned long long ga = (unsigned long long)(size_t)gsrc;
    tdm_u32x4 g0;
    g0[0] = 1u;                                        // count=1 (user mode)
    g0[1] = lds_off;                                   // lds_addr (bytes)
    g0[2] = (unsigned)(ga & 0xFFFFFFFFu);              // global_addr[31:0]
    g0[3] = (unsigned)((ga >> 32) & 0x01FFFFFFu)       // global_addr[56:32]
            | (2u << 30);                              // type=2 ("image")
    tdm_i32x8 g1;
    g1[0] = (int)((1u << 16)      // data_size = 2 bytes
                | (1u << 20)      // pad_enable
                | (3u << 22)      // pad_interval: 16 DWORDs (64B row)
                | (3u << 25));    // pad_amount:   4 DWORDs (16B pad)
    g1[1] = (int)(256u << 16);                         // tensor_dim0 = 256
    g1[2] = (int)(((unsigned)n_rows_tensor & 0xFFFFu) << 16); // tensor_dim1
    g1[3] = (int)(32u << 16);                          // tile_dim0 = 32
    g1[4] = (int)128;                                  // tile_dim1 = 128
    g1[5] = (int)256;                                  // tensor_dim0_stride
    g1[6] = 0;
    g1[7] = 0;
    tdm_i32x4 z4 = {0, 0, 0, 0};
#if __clang_major__ >= 23
    tdm_i32x8 z8 = {0, 0, 0, 0, 0, 0, 0, 0};
    __builtin_amdgcn_tensor_load_to_lds(g0, g1, z4, z4, z8, 0);
#else
    __builtin_amdgcn_tensor_load_to_lds(g0, g1, z4, z4, 0);
#endif
}
#endif // USE_TDM

// ---------------------------------------------------------------------------
// Weight prep: src (K=256 rows, Nc cols, row-major fp32)  ->
//   dst_hi/dst_lo transposed [rowOff+n][256] bf16 (hi / residual-lo split)
// ---------------------------------------------------------------------------
__global__ __launch_bounds__(256) void prep_weight(
    const float* __restrict__ src, unsigned short* __restrict__ dh,
    unsigned short* __restrict__ dl, int Nc, int rowOff)
{
    int id = blockIdx.x * 256 + threadIdx.x;       // Nc*256 total elements
    int n = id % Nc;
    int k = id / Nc;
    float x = src[(size_t)k * Nc + n];
    unsigned short h = bf16_rne(x);
    float lo = x - bf16_to_f(h);
    size_t o = (size_t)(rowOff + n) * KDIM + k;
    dh[o] = h;
    dl[o] = bf16_rne(lo);
}

__global__ void concat_bias(const float* __restrict__ b_off,
                            const float* __restrict__ b_attn,
                            float* __restrict__ bqp)
{
    int t = threadIdx.x;
    if (t < 64)       bqp[t] = b_off[t];
    else if (t < 96)  bqp[t] = b_attn[t - 64];
}

// ---------------------------------------------------------------------------
// bf16x3 WMMA GEMM:  C[M][Nc] = A[M][256] (fp32) * W[256][Nc] + bias
//   W supplied pre-transposed/split: Wh/Wl = [Nc][256] bf16 (hi / lo)
//   Block tile 128(M) x 128(N), 8 waves of 32 (wave32), wave tile 32x64.
//   Per K=32 chunk: 24x v_wmma_f32_16x16x32_bf16 per wave (hi*hi+lo*hi+hi*lo).
//   W tiles staged by the Tensor Data Mover (wave 0), overlapped with the
//   VALU fp32->bf16 hi/lo conversion of the A tile by all waves.
//   mode 0: C row-major (ld = Nc).
//   mode 1: value-projection layout (B,NH,NV,HD): row=b*16384+nv, col=h*32+hd
// ---------------------------------------------------------------------------
#define LDS_STRIDE 40   // 32 bf16 + 8 pad (80B rows, 16B aligned, conflict-free)

__global__ __launch_bounds__(256) void gemm_bf16x3(
    const float* __restrict__ A, const unsigned short* __restrict__ Wh,
    const unsigned short* __restrict__ Wl, const float* __restrict__ bias,
    float* __restrict__ C, int Ncols, int mode)
{
    __shared__ __align__(16) unsigned short As_hi[128 * LDS_STRIDE];
    __shared__ __align__(16) unsigned short As_lo[128 * LDS_STRIDE];
    __shared__ __align__(16) unsigned short Bs_hi[128 * LDS_STRIDE];
    __shared__ __align__(16) unsigned short Bs_lo[128 * LDS_STRIDE];

    const int t    = threadIdx.x;
    const int lane = t & 31;
    const int wid  = t >> 5;
    const int hh   = lane >> 4;    // lane half (0/1)
    const int lm   = lane & 15;
    const int wm   = wid & 3;      // wave M index (4)
    const int wn   = wid >> 2;     // wave N index (2)

    const long row0 = (long)blockIdx.y * 128;
    const int  n0   = blockIdx.x * 128;

#if USE_TDM
    const unsigned bhi_off = (unsigned)(size_t)(const void*)Bs_hi;
    const unsigned blo_off = (unsigned)(size_t)(const void*)Bs_lo;
#endif

    v8f acc[2][4];
#pragma unroll
    for (int i = 0; i < 2; ++i)
#pragma unroll
        for (int j = 0; j < 4; ++j) acc[i][j] = (v8f){};

    for (int kc = 0; kc < KDIM / 32; ++kc) {
        // ---- stage W chunk (128 n-rows x 32 k, pre-split bf16) ----
#if USE_TDM
        if (wid == 0) {   // one wave drives the DMA; EXEC ignored by TDM
            tdm_load_tile(Wh + (size_t)n0 * KDIM + kc * 32, bhi_off, Ncols);
            tdm_load_tile(Wl + (size_t)n0 * KDIM + kc * 32, blo_off, Ncols);
        }
#else
#pragma unroll
        for (int i = 0; i < 2; ++i) {
            int idx  = i * 256 + t;
            int rowN = idx >> 2;       // 0..127
            int c    = idx & 3;        // 16B chunk within row
            int n    = n0 + rowN;
            uint4 dh = {0, 0, 0, 0}, dl = {0, 0, 0, 0};
            if (n < Ncols) {
                size_t o = (size_t)n * KDIM + kc * 32 + c * 8;
                dh = *(const uint4*)(Wh + o);
                dl = *(const uint4*)(Wl + o);
            }
            *(uint4*)(Bs_hi + rowN * LDS_STRIDE + c * 8) = dh;
            *(uint4*)(Bs_lo + rowN * LDS_STRIDE + c * 8) = dl;
        }
#endif
        // ---- stage A chunk (128 rows x 32 k, fp32 -> bf16 hi/lo) ----
#pragma unroll
        for (int i = 0; i < 4; ++i) {
            int idx = i * 256 + t;
            int row = idx >> 3;        // 0..127
            int c4  = idx & 7;         // float4 within row
            const float4 v = *(const float4*)(A + (row0 + row) * KDIM +
                                              kc * 32 + c4 * 4);
            unsigned short h0 = bf16_rne(v.x), h1 = bf16_rne(v.y),
                           h2 = bf16_rne(v.z), h3 = bf16_rne(v.w);
            unsigned short l0 = bf16_rne(v.x - bf16_to_f(h0));
            unsigned short l1 = bf16_rne(v.y - bf16_to_f(h1));
            unsigned short l2 = bf16_rne(v.z - bf16_to_f(h2));
            unsigned short l3 = bf16_rne(v.w - bf16_to_f(h3));
            uint2 ph = { (unsigned)h0 | ((unsigned)h1 << 16),
                         (unsigned)h2 | ((unsigned)h3 << 16) };
            uint2 pl = { (unsigned)l0 | ((unsigned)l1 << 16),
                         (unsigned)l2 | ((unsigned)l3 << 16) };
            *(uint2*)(As_hi + row * LDS_STRIDE + c4 * 4) = ph;
            *(uint2*)(As_lo + row * LDS_STRIDE + c4 * 4) = pl;
        }
#if USE_TDM
        if (wid == 0) __builtin_amdgcn_s_wait_tensorcnt(0);
#endif
        __syncthreads();

        // ---- load fragments per the 16-bit WMMA VGPR layouts ----
        Frag16 ah[2], al[2], bh[4], bl[4];
#pragma unroll
        for (int i = 0; i < 2; ++i) {
            const unsigned short* pr = As_hi + (wm * 32 + i * 16 + lm) * LDS_STRIDE;
            const unsigned short* pl = As_lo + (wm * 32 + i * 16 + lm) * LDS_STRIDE;
            ah[i].f4[0] = *(const float4*)(pr + 8 * hh);        // K = e + 8h
            ah[i].f4[1] = *(const float4*)(pr + 16 + 8 * hh);   // K = e-8+16+8h
            al[i].f4[0] = *(const float4*)(pl + 8 * hh);
            al[i].f4[1] = *(const float4*)(pl + 16 + 8 * hh);
        }
#pragma unroll
        for (int j = 0; j < 4; ++j) {
            const unsigned short* pr =
                Bs_hi + (wn * 64 + j * 16 + lm) * LDS_STRIDE; // K = e + 8h, e-8+16+8h
            const unsigned short* pl =
                Bs_lo + (wn * 64 + j * 16 + lm) * LDS_STRIDE;
            bh[j].f4[0] = *(const float4*)(pr + 8 * hh);
            bh[j].f4[1] = *(const float4*)(pr + 16 + 8 * hh);
            bl[j].f4[0] = *(const float4*)(pl + 8 * hh);
            bl[j].f4[1] = *(const float4*)(pl + 16 + 8 * hh);
        }

        // ---- bf16x3 MACs: hi*hi + lo*hi + hi*lo (fp32 accumulate) ----
#pragma unroll
        for (int i = 0; i < 2; ++i)
#pragma unroll
            for (int j = 0; j < 4; ++j) {
                acc[i][j] = __builtin_amdgcn_wmma_f32_16x16x32_bf16(
                    false, ah[i].v, false, bh[j].v, (short)0, acc[i][j], false, false);
                acc[i][j] = __builtin_amdgcn_wmma_f32_16x16x32_bf16(
                    false, al[i].v, false, bh[j].v, (short)0, acc[i][j], false, false);
                acc[i][j] = __builtin_amdgcn_wmma_f32_16x16x32_bf16(
                    false, ah[i].v, false, bl[j].v, (short)0, acc[i][j], false, false);
            }
        __syncthreads();
    }

    // ---- epilogue: bias + scatter per output mode ----
#pragma unroll
    for (int i = 0; i < 2; ++i) {
#pragma unroll
        for (int j = 0; j < 4; ++j) {
            int gn = n0 + wn * 64 + j * 16 + lm;
            if (gn < Ncols) {
                float bv = bias[gn];
#pragma unroll
                for (int r = 0; r < 8; ++r) {
                    long gm = row0 + wm * 32 + i * 16 + r + 8 * hh; // C layout: M=r+8h
                    float val = acc[i][j][r] + bv;
                    size_t oidx;
                    if (mode == 0) {
                        oidx = (size_t)gm * Ncols + gn;
                    } else { // (B,NH,NV,HD) head-major value layout
                        int bb = (int)(gm >> 14);
                        int nv = (int)(gm & 16383);
                        int hd_head = gn >> 5;
                        int hd = gn & 31;
                        oidx = (((size_t)(bb * NHEAD + hd_head) << 14) + nv) * HDIM + hd;
                    }
                    C[oidx] = val;
                }
            }
        }
    }
}

// ---------------------------------------------------------------------------
// Deformable sampling: one query per wave, lane = head-dim channel.
//   qp  : (B*N, 96) fp32  [64 offset logits | 32 attn logits], bias included
//   ref : (B*N, 2)
//   vpr : (B, NH, NV, HD) head-major projected values
//   outw: (B*N, 256) weighted samples
// ---------------------------------------------------------------------------
__global__ __launch_bounds__(256) void deform_sample(
    const float* __restrict__ qp, const float* __restrict__ ref,
    const float* __restrict__ vpr, float* __restrict__ outw)
{
    __shared__ float sqp[8][96];
    __shared__ float sref[8][2];

    const int t    = threadIdx.x;
    const int wid  = t >> 5;
    const int lane = t & 31;
    const size_t q = (size_t)blockIdx.x * 8 + wid;

#pragma unroll
    for (int i = 0; i < 3; ++i)
        sqp[wid][lane + 32 * i] = qp[q * 96 + lane + 32 * i];
    if (lane < 2) sref[wid][lane] = ref[q * 2 + lane];
    __syncthreads();

    const int  b  = (int)(q >> 14);
    const float rx = sref[wid][0];
    const float ry = sref[wid][1];

    for (int h = 0; h < NHEAD; ++h) {
        // softmax over P=4 (broadcast LDS reads, redundant per lane)
        float lg[NPTS];
#pragma unroll
        for (int p = 0; p < NPTS; ++p) lg[p] = sqp[wid][64 + h * NPTS + p];
        float m = fmaxf(fmaxf(lg[0], lg[1]), fmaxf(lg[2], lg[3]));
        float e[NPTS];
        float s = 0.f;
#pragma unroll
        for (int p = 0; p < NPTS; ++p) { e[p] = __expf(lg[p] - m); s += e[p]; }
        const float inv = 1.0f / s;

        const float* vh = vpr + ((size_t)(b * NHEAD + h) << 19); // *NV*HD
        float acc = 0.f;
#pragma unroll
        for (int p = 0; p < NPTS; ++p) {
            float ox = sqp[wid][(h * NPTS + p) * 2 + 0];
            float oy = sqp[wid][(h * NPTS + p) * 2 + 1];
            // loc=(ref+off)*2-1; unnorm=((loc+1)*HW-1)/2; round half-to-even
            float ix = rintf((rx + ox) * (float)HW - 0.5f);
            float iy = rintf((ry + oy) * (float)HW - 0.5f);
            if (ix >= 0.f && ix < (float)HW && iy >= 0.f && iy < (float)HW) {
                int fl = (int)iy * HW + (int)ix;
                acc += (e[p] * inv) * vh[(size_t)fl * HDIM + lane];
            }
        }
        outw[q * 256 + h * HDIM + lane] = acc;
    }
}

// ---------------------------------------------------------------------------
// Host-side launch
// ---------------------------------------------------------------------------
extern "C" void kernel_launch(void* const* d_in, const int* in_sizes, int n_in,
                              void* d_out, int out_size, void* d_ws, size_t ws_size,
                              hipStream_t stream) {
    const float* query  = (const float*)d_in[0];
    const float* refp   = (const float*)d_in[1];
    const float* value  = (const float*)d_in[2];
    const float* W_off  = (const float*)d_in[3];
    const float* b_off  = (const float*)d_in[4];
    const float* W_attn = (const float*)d_in[5];
    const float* b_attn = (const float*)d_in[6];
    const float* W_v    = (const float*)d_in[7];
    const float* b_v    = (const float*)d_in[8];
    const float* W_out  = (const float*)d_in[9];
    const float* b_out  = (const float*)d_in[10];
    float* out = (float*)d_out;

    size_t off = 0;
    char*  ws  = (char*)d_ws;
    auto alloc = [&](size_t bytes) -> void* {
        void* p = ws + off;
        off += (bytes + 255) & ~(size_t)255;
        return p;
    };
    unsigned short* Wv_hi  = (unsigned short*)alloc((size_t)KDIM * KDIM * 2);
    unsigned short* Wv_lo  = (unsigned short*)alloc((size_t)KDIM * KDIM * 2);
    unsigned short* Wo_hi  = (unsigned short*)alloc((size_t)KDIM * KDIM * 2);
    unsigned short* Wo_lo  = (unsigned short*)alloc((size_t)KDIM * KDIM * 2);
    unsigned short* Wqp_hi = (unsigned short*)alloc((size_t)96 * KDIM * 2);
    unsigned short* Wqp_lo = (unsigned short*)alloc((size_t)96 * KDIM * 2);
    float* bqp   = (float*)alloc(96 * 4);
    float* qproj = (float*)alloc((size_t)MROWS * 96 * 4);
    float* vproj = (float*)alloc((size_t)MROWS * KDIM * 4);
    float* wres  = (float*)alloc((size_t)MROWS * KDIM * 4);

    // 1) weight prep: transpose + bf16 hi/lo split
    prep_weight<<<256, 256, 0, stream>>>(W_v,    Wv_hi,  Wv_lo,  256, 0);
    prep_weight<<<256, 256, 0, stream>>>(W_out,  Wo_hi,  Wo_lo,  256, 0);
    prep_weight<<<64,  256, 0, stream>>>(W_off,  Wqp_hi, Wqp_lo, 64,  0);
    prep_weight<<<32,  256, 0, stream>>>(W_attn, Wqp_hi, Wqp_lo, 32,  64);
    concat_bias<<<1, 128, 0, stream>>>(b_off, b_attn, bqp);

    // 2) GEMMs (bf16x3 WMMA, TDM-staged weights) + sampling
    dim3 blk(256);
    gemm_bf16x3<<<dim3(1, MROWS / 128), blk, 0, stream>>>(
        query, Wqp_hi, Wqp_lo, bqp, qproj, 96, 0);
    gemm_bf16x3<<<dim3(2, MROWS / 128), blk, 0, stream>>>(
        value, Wv_hi, Wv_lo, b_v, vproj, 256, 1);
    deform_sample<<<MROWS / 8, 256, 0, stream>>>(qproj, refp, vproj, wres);
    gemm_bf16x3<<<dim3(2, MROWS / 128), blk, 0, stream>>>(
        wres, Wo_hi, Wo_lo, b_out, out, 256, 0);
}
